// MultiHeadCrossAttention_3539053052041
// MI455X (gfx1250) — matrix operand order repaired and checked by reading, not compiled, not run on hardware
//
#include <hip/hip_runtime.h>
#include <hip/hip_bf16.h>
#include <stdint.h>

// Problem constants (fixed by the reference)
#define B_   4
#define S_   2048
#define D_   1024
#define H_   16
#define HD_  64
#define DM_  3072          // mixed width = 3*D
// softmax is computed base-2: p = exp2(s * C1 + madd), with
// C1 = log2(e)/sqrt(hd) and madd = NEG_BIG*log2(e)/sqrt(hd) for masked entries.
#define C1_     (0.18033688011f)     // 0.125 * log2(e)
#define MADD2_  (-1.8033688e10f)     // -1e11 * 0.125 * log2(e)

typedef __attribute__((ext_vector_type(16))) __bf16 bf16x16;
typedef __attribute__((ext_vector_type(8)))  __bf16 bf16x8;
typedef __attribute__((ext_vector_type(4)))  __bf16 bf16x4;
typedef __attribute__((ext_vector_type(2)))  __bf16 bf16x2;
typedef __attribute__((ext_vector_type(8)))  float  f32x8;

static __device__ __forceinline__ f32x8 wmma_bf16(bf16x16 a, bf16x16 b, f32x8 c) {
  // D = A(16x32 bf16) * B(32x16 bf16) + C(16x16 f32)
  return __builtin_amdgcn_wmma_f32_16x16x32_bf16(false, a, false, b, (short)0, c,
                                                 false, false);
}

// A-operand (16x32 bf16) fragment from row-major [row][k] storage.
// Lane l: row = l%16; K elems {0..7,16..23} (l<16) or {8..15,24..31} (l>=16).
static __device__ __forceinline__ bf16x16 load_a_frag(const __bf16* base, int stride, int lane) {
  const __bf16* p = base + (size_t)(lane & 15) * stride + ((lane >> 4) << 3);
  bf16x8 lo = *(const bf16x8*)p;          // 16B load
  bf16x8 hi = *(const bf16x8*)(p + 16);   // 16B load
  return __builtin_shufflevector(lo, hi, 0,1,2,3,4,5,6,7,8,9,10,11,12,13,14,15);
}

// B-operand (32x16 bf16) fragment where B[k][n] = M[n][k] with M row-major [n][k].
// Lane l: col n = l%16; K elems {0..7,16..23} (l<16) or {8..15,24..31} (l>=16).
static __device__ __forceinline__ bf16x16 load_b_frag(const __bf16* base, int stride, int lane) {
  const __bf16* p = base + (size_t)(lane & 15) * stride + ((lane >> 4) << 3);
  bf16x8 lo = *(const bf16x8*)p;          // 16B load
  bf16x8 hi = *(const bf16x8*)(p + 16);   // 16B load
  return __builtin_shufflevector(lo, hi, 0,1,2,3,4,5,6,7,8,9,10,11,12,13,14,15);
}

// ---------------- conversion kernels ----------------
__global__ __launch_bounds__(256) void cvt_f32_to_bf16(const float* __restrict__ src,
                                                       __bf16* __restrict__ dst, int n4) {
  int idx = blockIdx.x * blockDim.x + threadIdx.x;
  if (idx >= n4) return;
  int i = idx << 2;
  float4 f = *(const float4*)(src + i);
  bf16x4 o = { (__bf16)f.x, (__bf16)f.y, (__bf16)f.z, (__bf16)f.w };
  *(bf16x4*)(dst + i) = o;
}

// dec_input -> mixed[:, :, 2048:3072] (bf16)
__global__ __launch_bounds__(256) void cvt_dec_to_mixed(const float* __restrict__ dec,
                                                        __bf16* __restrict__ mixed) {
  int idx = blockIdx.x * blockDim.x + threadIdx.x;   // over B*S*D/4
  int i = idx << 2;
  int row = i >> 10;          // /D_
  int col = i & (D_ - 1);
  float4 f = *(const float4*)(dec + i);
  bf16x4 o = { (__bf16)f.x, (__bf16)f.y, (__bf16)f.z, (__bf16)f.w };
  *(bf16x4*)(mixed + (size_t)row * DM_ + 2048 + col) = o;
}

// maskT[k*S + q] = mask[q*S + k] ? MADD2 : 0  (tiled transpose via LDS)
__global__ __launch_bounds__(256) void build_maskT(const uint8_t* __restrict__ mask,
                                                   float* __restrict__ maskT) {
  __shared__ uint8_t tile[64][80];   // [q][k], padded row stride (16B aligned)
  const int qt = blockIdx.y * 64, kt = blockIdx.x * 64;
  {
    const int r  = threadIdx.x >> 2;          // 0..63 (q)
    const int c0 = (threadIdx.x & 3) << 4;    // 0,16,32,48 (k)
    uint4 d = *(const uint4*)(mask + (size_t)(qt + r) * S_ + kt + c0);
    *(uint4*)&tile[r][c0] = d;
  }
  __syncthreads();
  const int k2 = threadIdx.x >> 2;            // 0..63 (k)
  const int q0 = (threadIdx.x & 3) << 4;      // 0,16,32,48 (q)
  float* dst = maskT + (size_t)(kt + k2) * S_ + qt + q0;
#pragma unroll
  for (int j = 0; j < 16; j += 4) {
    float4 o;
    o.x = tile[q0 + j + 0][k2] ? MADD2_ : 0.0f;
    o.y = tile[q0 + j + 1][k2] ? MADD2_ : 0.0f;
    o.z = tile[q0 + j + 2][k2] ? MADD2_ : 0.0f;
    o.w = tile[q0 + j + 3][k2] ? MADD2_ : 0.0f;
    *(float4*)(dst + j) = o;
  }
}

// ---------------- GEMM: C[M,N] = A[M,K] * W[N,K]^T + bias ----------------
// WG = 256 threads = 8 waves arranged 2(M) x 4(N); wave tile 64x32 (8 accum frags).
template <bool OUT_BF16>
__global__ __launch_bounds__(256) void gemm_bf16_nt(const __bf16* __restrict__ A, int lda,
                                                    const __bf16* __restrict__ W, int ldw,
                                                    const float* __restrict__ bias,
                                                    void* __restrict__ Cout, int ldc,
                                                    int Ktot) {
  const int lane = threadIdx.x & 31;
  const int wave = threadIdx.x >> 5;
  const int m0 = blockIdx.y * 128 + (wave >> 2) * 64;
  const int n0 = blockIdx.x * 128 + (wave & 3) * 32;

  f32x8 acc[4][2] = {};
  for (int kk = 0; kk < Ktot; kk += 32) {
    bf16x16 af[4], bf_[2];
#pragma unroll
    for (int i = 0; i < 4; ++i)
      af[i] = load_a_frag(A + (size_t)(m0 + 16 * i) * lda + kk, lda, lane);
#pragma unroll
    for (int j = 0; j < 2; ++j)
      bf_[j] = load_b_frag(W + (size_t)(n0 + 16 * j) * ldw + kk, ldw, lane);
#pragma unroll
    for (int i = 0; i < 4; ++i)
#pragma unroll
      for (int j = 0; j < 2; ++j)
        acc[i][j] = wmma_bf16(af[i], bf_[j], acc[i][j]);
  }

#pragma unroll
  for (int j = 0; j < 2; ++j) {
    const int col = n0 + 16 * j + (lane & 15);
    const float bv = bias[col];
#pragma unroll
    for (int i = 0; i < 4; ++i) {
#pragma unroll
      for (int v = 0; v < 8; ++v) {
        const int row = m0 + 16 * i + v + ((lane >> 4) << 3);
        const float val = acc[i][j][v] + bv;
        if (OUT_BF16)
          ((__bf16*)Cout)[(size_t)row * ldc + col] = (__bf16)val;
        else
          ((float*)Cout)[(size_t)row * ldc + col] = val;
      }
    }
  }
}

// ---------------- flash attention (max-free softmax, WMMA row-sums) ----------------
// grid = (S/128, H, B), block = 128 (4 waves). Each wave owns 32 queries (2 M-frags).
// Key-slot interleave: score frag0 col c <-> key kb+2c, frag1 col c <-> key kb+2c+1
// (packs the two per-row probabilities into one 32-bit LDS store; slot<->key stays
// identity for the P tile and Vt). Row sums are accumulated with an extra WMMA
// against an all-ones B operand -- no cross-lane reductions anywhere.
// Output written directly in the reference's "no-transpose reshape" layout:
// values[b, h*128 + q/16, (q%16)*64 + f].
__global__ __launch_bounds__(128) void attn_flash(const __bf16* __restrict__ mixed,
                                                  const float* __restrict__ maskT,
                                                  __bf16* __restrict__ values) {
  __shared__ alignas(32) __bf16 Kl[32 * HD_];        // [key_local][feat]
  __shared__ alignas(32) __bf16 Vt[HD_ * 32];        // [feat][key_local]
  __shared__ alignas(32) __bf16 Pb[4 * 2 * 16 * 32]; // per-wave P tiles [mt][q16][slot32]

  const int lane = threadIdx.x & 31;
  const int wave = threadIdx.x >> 5;
  const int b = blockIdx.z, h = blockIdx.y;
  const int qbase = blockIdx.x * 128 + wave * 32;    // first of 32 query rows
  const int qcol = h * 192, kcol = qcol + 64, vcol = qcol + 128;

  // preload Q fragments for both 16-row M-tiles
  bf16x16 qa[2][2];
#pragma unroll
  for (int mt = 0; mt < 2; ++mt) {
    const __bf16* qptr = mixed + ((size_t)(b * S_ + qbase + mt * 16)) * DM_ + qcol;
    qa[mt][0] = load_a_frag(qptr,      DM_, lane);   // hd 0..31
    qa[mt][1] = load_a_frag(qptr + 32, DM_, lane);   // hd 32..63
  }

  bf16x16 onesB;
#pragma unroll
  for (int i = 0; i < 16; ++i) onesB[i] = (__bf16)1.0f;

  f32x8 o[2][4] = {};     // O accumulators, [mt][feat-tile]
  f32x8 L[2]    = {};     // row-sum accumulators

  __bf16* Pw = Pb + wave * (2 * 16 * 32);
  const int c = lane & 15;
  const int skey = threadIdx.x >> 2;                // K staging: key 0..31
  const int sf0  = (threadIdx.x & 3) << 4;          // K staging: feat base 0,16,32,48
  const int vkey = (threadIdx.x >> 3) << 1;         // V staging: key pair 0,2,..,30
  const int vf0  = (threadIdx.x & 7) << 3;          // V staging: feat base 0,8,..,56

  for (int kb = 0; kb < S_; kb += 32) {
    __syncthreads();  // previous iteration's Kl/Vt readers are done
    {   // stage K: plain row copy, 2x b128 in / 2x b128 out
      const __bf16* srcK = mixed + ((size_t)(b * S_ + kb + skey)) * DM_ + kcol + sf0;
      bf16x8 k0 = ((const bf16x8*)srcK)[0];
      bf16x8 k1 = ((const bf16x8*)srcK)[1];
      ((bf16x8*)(Kl + skey * HD_ + sf0))[0] = k0;
      ((bf16x8*)(Kl + skey * HD_ + sf0))[1] = k1;
    }
    {   // stage V^T: adjacent key pair per thread -> 8 packed 32-bit LDS stores
      const __bf16* srcV = mixed + ((size_t)(b * S_ + kb + vkey)) * DM_ + vcol + vf0;
      bf16x8 va = *(const bf16x8*)srcV;
      bf16x8 vb = *(const bf16x8*)(srcV + DM_);
#pragma unroll
      for (int j = 0; j < 8; ++j) {
        bf16x2 pr; pr[0] = va[j]; pr[1] = vb[j];
        *(bf16x2*)(Vt + (vf0 + j) * 32 + vkey) = pr;
      }
    }
    __syncthreads();

    const bf16x16 bk00 = load_b_frag(Kl,            2 * HD_, lane);  // even keys, hd lo
    const bf16x16 bk01 = load_b_frag(Kl + 32,       2 * HD_, lane);  // even keys, hd hi
    const bf16x16 bk10 = load_b_frag(Kl + HD_,      2 * HD_, lane);  // odd keys, hd lo
    const bf16x16 bk11 = load_b_frag(Kl + HD_ + 32, 2 * HD_, lane);  // odd keys, hd hi

#pragma unroll
    for (int mt = 0; mt < 2; ++mt) {
      // scores 16(q) x 32(keys): frag0 = even keys, frag1 = odd keys
      f32x8 s0 = {}, s1 = {};
      s0 = wmma_bf16(qa[mt][0], bk00, s0);
      s0 = wmma_bf16(qa[mt][1], bk01, s0);
      s1 = wmma_bf16(qa[mt][0], bk10, s1);
      s1 = wmma_bf16(qa[mt][1], bk11, s1);

      // additive mask (pre-scaled by log2(e)/8): 8 consecutive q per vector load
      const float* mp = maskT + (size_t)(kb + 2 * c) * S_
                      + qbase + mt * 16 + ((lane >> 4) << 3);
      const f32x8 madd0 = *(const f32x8*)mp;         // key kb+2c
      const f32x8 madd1 = *(const f32x8*)(mp + S_);  // key kb+2c+1

      __bf16* Pt = Pw + mt * (16 * 32);
#pragma unroll
      for (int v = 0; v < 8; ++v) {
        const float p0 = __builtin_amdgcn_exp2f(s0[v] * C1_ + madd0[v]);
        const float p1 = __builtin_amdgcn_exp2f(s1[v] * C1_ + madd1[v]);
        const int r = v + ((lane >> 4) << 3);
        bf16x2 pk; pk[0] = (__bf16)p0; pk[1] = (__bf16)p1;   // slots 2c, 2c+1
        *(bf16x2*)(Pt + r * 32 + 2 * c) = pk;
      }
    }
    // same-wave LDS RAW: HW keeps DS ops in order; stop compiler reordering only.
    __builtin_amdgcn_wave_barrier();

    const bf16x16 bv0 = load_b_frag(Vt,           32, lane);
    const bf16x16 bv1 = load_b_frag(Vt + 16 * 32, 32, lane);
    const bf16x16 bv2 = load_b_frag(Vt + 32 * 32, 32, lane);
    const bf16x16 bv3 = load_b_frag(Vt + 48 * 32, 32, lane);
#pragma unroll
    for (int mt = 0; mt < 2; ++mt) {
      const bf16x16 pa = load_a_frag(Pw + mt * (16 * 32), 32, lane);
      L[mt]    = wmma_bf16(pa, onesB, L[mt]);   // row sums via matrix engine
      o[mt][0] = wmma_bf16(pa, bv0, o[mt][0]);
      o[mt][1] = wmma_bf16(pa, bv1, o[mt][1]);
      o[mt][2] = wmma_bf16(pa, bv2, o[mt][2]);
      o[mt][3] = wmma_bf16(pa, bv3, o[mt][3]);
    }
  }

  // normalize and store in "values" layout
#pragma unroll
  for (int mt = 0; mt < 2; ++mt) {
    const size_t s2 = (size_t)h * 128 + ((qbase + mt * 16) >> 4);
    __bf16* vrow = values + ((size_t)b * S_ + s2) * D_;
#pragma unroll
    for (int v = 0; v < 8; ++v) {
      const float inv = 1.0f / L[mt][v];
      const int r = v + ((lane >> 4) << 3);
      vrow[r * 64 +      c] = (__bf16)(o[mt][0][v] * inv);
      vrow[r * 64 + 16 + c] = (__bf16)(o[mt][1][v] * inv);
      vrow[r * 64 + 32 + c] = (__bf16)(o[mt][2][v] * inv);
      vrow[r * 64 + 48 + c] = (__bf16)(o[mt][3][v] * inv);
    }
  }
}

// ---------------- launch ----------------
extern "C" void kernel_launch(void* const* d_in, const int* in_sizes, int n_in,
                              void* d_out, int out_size, void* d_ws, size_t ws_size,
                              hipStream_t stream) {
  const float*   dec  = (const float*)d_in[0];
  const float*   enc  = (const float*)d_in[1];
  const uint8_t* mask = (const uint8_t*)d_in[2];   // jnp.bool_ -> 1 byte/elem
  const float*   Wqk  = (const float*)d_in[3];     // [2048, 1024]
  const float*   bqk  = (const float*)d_in[4];     // [2048]
  const float*   Wl   = (const float*)d_in[5];     // [1024, 1024]
  const float*   bl   = (const float*)d_in[6];     // [1024]
  float* out = (float*)d_out;

  // workspace layout
  const size_t szEnc   = (size_t)B_ * S_ * D_ * 2;     // 16 MB bf16
  const size_t szMixed = (size_t)B_ * S_ * DM_ * 2;    // 48 MB bf16
  const size_t szWqk   = (size_t)2048 * 1024 * 2;      //  4 MB bf16
  const size_t szWl    = (size_t)1024 * 1024 * 2;      //  2 MB bf16
  const size_t szVals  = (size_t)B_ * S_ * D_ * 2;     // 16 MB bf16
  const size_t szMaskT = (size_t)S_ * S_ * 4;          // 16 MB f32
  size_t off = 0;
  char* ws = (char*)d_ws;
  auto take = [&](size_t bytes) { void* p = ws + off; off += (bytes + 255) & ~(size_t)255; return p; };
  __bf16* encB   = (__bf16*)take(szEnc);
  __bf16* mixedB = (__bf16*)take(szMixed);
  __bf16* WqkB   = (__bf16*)take(szWqk);
  __bf16* WlB    = (__bf16*)take(szWl);
  __bf16* valsB  = (__bf16*)take(szVals);
  float*  maskT  = (float*)take(szMaskT);
  if (off > ws_size) return;  // insufficient scratch (shouldn't happen)

  // 1) precision conversions + mask preprocessing
  cvt_f32_to_bf16<<<(B_*S_*D_/4 + 255)/256, 256, 0, stream>>>(enc, encB, B_*S_*D_/4);
  cvt_f32_to_bf16<<<(2048*1024/4 + 255)/256, 256, 0, stream>>>(Wqk, WqkB, 2048*1024/4);
  cvt_f32_to_bf16<<<(1024*1024/4 + 255)/256, 256, 0, stream>>>(Wl,  WlB,  1024*1024/4);
  cvt_dec_to_mixed<<<B_*S_*D_/4/256, 256, 0, stream>>>(dec, mixedB);
  build_maskT<<<dim3(S_/64, S_/64), 256, 0, stream>>>(mask, maskT);

  // 2) QK projection into mixed[:, :, 0:2048]
  gemm_bf16_nt<true><<<dim3(2048/128, (B_*S_)/128), 256, 0, stream>>>(
      encB, D_, WqkB, D_, bqk, (void*)mixedB, DM_, D_);

  // 3) flash attention -> valsB in reshaped layout
  attn_flash<<<dim3(S_/128, H_, B_), 128, 0, stream>>>(mixedB, maskT, valsB);

  // 4) output linear -> d_out (f32)
  gemm_bf16_nt<false><<<dim3(1024/128, (B_*S_)/128), 256, 0, stream>>>(
      valsB, D_, WlB, D_, bl, (void*)out, D_, D_);
}
